// DynamicPatchAttacker_21620865368361
// MI455X (gfx1250) — matrix operand order repaired and checked by reading, not compiled, not run on hardware
//
#include <hip/hip_runtime.h>
#include <hip/hip_bf16.h>

// ---------------------------------------------------------------------------
// DynamicPatchAttacker for MI455X (gfx1250, wave32).
//  - NMS: 1 workgroup/image, bitonic sort + greedy suppression.
//  - 16 sequential steps; per step: generator kernel (WMMA f16 convs) + paste.
//  - conv2 / conv3 run on v_wmma_f32_16x16x32_f16 (K=32 per 3x3 tap, 9 taps
//    chained through the f32 accumulator). conv1 (K=27) is scalar f32.
// ---------------------------------------------------------------------------

typedef __attribute__((ext_vector_type(16))) _Float16 v16h;
typedef __attribute__((ext_vector_type(8)))  float    v8f;
typedef __attribute__((ext_vector_type(2)))  _Float16 h2v;

#define B_   8
#define IMG_ 512
#define NC_  1024
#define MB_  16
#define P_   128

#define TILE 16
#define RIN  22   // input region  (TILE + 6 halo)
#define RH1  20   // h1 region     (TILE + 4)
#define RH2  18   // h2 region     (TILE + 2)

static __device__ __forceinline__ int imin(int a, int b) { return a < b ? a : b; }

// ---------------------------------------------------------------------------
// init copy: images -> d_out
// ---------------------------------------------------------------------------
__global__ __launch_bounds__(256) void copy_kernel(const float* __restrict__ src,
                                                   float* __restrict__ dst, int n) {
  int i = blockIdx.x * 256 + threadIdx.x;
  if (i < n) dst[i] = src[i];
}

// ---------------------------------------------------------------------------
// NMS: one block (256 thr) per image.
// ---------------------------------------------------------------------------
__global__ __launch_bounds__(256) void nms_kernel(const float* __restrict__ boxes,
                                                  const float* __restrict__ scores,
                                                  float* __restrict__ wsBoxes,
                                                  int* __restrict__ wsValid,
                                                  float* __restrict__ wsMaxSc,
                                                  float* __restrict__ wsLoss) {
  const int img = blockIdx.x;
  const int tid = threadIdx.x;

  __shared__ float skey[NC_];
  __shared__ int   sidx[NC_];
  __shared__ float sy0[NC_], sx0[NC_], sy1[NC_], sx1[NC_];
  __shared__ unsigned char sval[NC_], skept[NC_];
  __shared__ int hit;

  for (int i = tid; i < NC_; i += 256) { skey[i] = scores[img * NC_ + i]; sidx[i] = i; }
  __syncthreads();

  // bitonic sort, descending score, stable (tie -> smaller original index first)
  for (int k = 2; k <= NC_; k <<= 1) {
    for (int j = k >> 1; j > 0; j >>= 1) {
      for (int i = tid; i < NC_; i += 256) {
        int ixj = i ^ j;
        if (ixj > i) {
          bool up = ((i & k) == 0);
          float a = skey[i], b = skey[ixj];
          int ia = sidx[i], ib = sidx[ixj];
          bool aFirst = (a > b) || (a == b && ia < ib);   // desired order
          bool doSwap = up ? !aFirst : aFirst;
          if (doSwap) { skey[i] = b; skey[ixj] = a; sidx[i] = ib; sidx[ixj] = ia; }
        }
      }
      __syncthreads();
    }
  }

  // gather sorted boxes + validity
  for (int i = tid; i < NC_; i += 256) {
    int src = sidx[i];
    const float* bp = boxes + (size_t)(img * NC_ + src) * 4;
    float y0 = bp[0], x0 = bp[1], y1 = bp[2], x1 = bp[3];
    sy0[i] = y0; sx0[i] = x0; sy1[i] = y1; sx1[i] = x1;
    float bh = y1 - y0, bw = x1 - x0;
    float sc = skey[i];
    sval[i] = (bh * bw > 100.0f) && (sc >= 0.4f) && (bh <= 512.0f) && (bw <= 512.0f);
    skept[i] = 0;
  }
  __syncthreads();

  // greedy suppression scan (serial in i, parallel over prior kept boxes)
  for (int i = 0; i < NC_; ++i) {
    if (tid == 0) hit = 0;
    __syncthreads();
    if (sval[i]) {
      float ay0 = sy0[i], ax0 = sx0[i], ay1 = sy1[i], ax1 = sx1[i];
      float aarea = (ay1 - ay0) * (ax1 - ax0);
      int local = 0;
      for (int j = tid; j < i; j += 256) {
        if (skept[j]) {
          float iy = fmaxf(fminf(ay1, sy1[j]) - fmaxf(ay0, sy0[j]), 0.0f);
          float ix = fmaxf(fminf(ax1, sx1[j]) - fmaxf(ax0, sx0[j]), 0.0f);
          float inter = iy * ix;
          float barea = (sy1[j] - sy0[j]) * (sx1[j] - sx0[j]);
          float iou = inter / fmaxf(aarea + barea - inter, 1e-6f);
          if (iou > 0.5f) local = 1;
        }
      }
      if (local) atomicOr(&hit, 1);
    }
    __syncthreads();
    if (tid == 0) skept[i] = (sval[i] && !hit) ? 1 : 0;
    __syncthreads();
  }

  if (tid == 0) {
    int cnt = 0; float msc = 0.0f;
    for (int i = 0; i < NC_ && cnt < MB_; ++i) {
      if (skept[i]) {
        float* ob = wsBoxes + (size_t)(img * MB_ + cnt) * 4;
        ob[0] = floorf(sy0[i]); ob[1] = floorf(sx0[i]);
        ob[2] = floorf(sy1[i]); ob[3] = floorf(sx1[i]);
        wsValid[img * MB_ + cnt] = 1;
        if (cnt == 0) msc = skey[i];
        ++cnt;
      }
    }
    for (; cnt < MB_; ++cnt) {
      float* ob = wsBoxes + (size_t)(img * MB_ + cnt) * 4;
      ob[0] = ob[1] = ob[2] = ob[3] = 0.0f;
      wsValid[img * MB_ + cnt] = 0;
    }
    wsMaxSc[img] = msc;
    wsLoss[img] = 0.0f;
  }
}

// ---------------------------------------------------------------------------
// Generator: one block per (tile, image). grid = (64, B). 256 thr = 8 waves.
// Dynamic LDS layout (bytes, 16B aligned blocks):
//   s_in  f32 [22][22][3]        5808
//   s_h1  f16 [20][20][32]      25600
//   s_h2  f16 [18][18][32]      20736
//   s_w2  f16 [9][32][32] (t,n,k) 18432
//   s_w3  f16 [9][16][32] (t,n,k)  9216
//   s_w1  f32 [27][32]           3456
//   s_b1/b2 f32[32]x2, s_b3 f32[4]  272
//   s_gen f32 [16][16][3]        3072
//   s_red f32 [8]                  32   -> total 86624 B (within 320KB/WGP)
// ---------------------------------------------------------------------------
#define GEN_SMEM_BYTES 86624

__global__ __launch_bounds__(256) void gen_kernel(const float* __restrict__ images,
                                                  const float* __restrict__ wsBoxes,
                                                  const int* __restrict__ wsValid,
                                                  const float* __restrict__ W1, const float* __restrict__ b1,
                                                  const float* __restrict__ W2, const float* __restrict__ b2,
                                                  const float* __restrict__ W3, const float* __restrict__ b3,
                                                  float* __restrict__ genbuf,
                                                  float* __restrict__ lossacc,
                                                  int step) {
  const int img = blockIdx.y;
  if (!wsValid[img * MB_ + step]) return;

  const int tile = blockIdx.x;
  const int ty = tile >> 3, tx = tile & 7;
  const int r0 = ty * TILE, c0 = tx * TILE;
  const int tid = threadIdx.x;
  const int lane = tid & 31;
  const int wave = tid >> 5;

  extern __shared__ char smem[];
  float*    s_in  = (float*)smem;                          // 22*22*3
  _Float16* s_h1  = (_Float16*)(s_in + RIN * RIN * 3);     // 20*20*32
  _Float16* s_h2  = s_h1 + RH1 * RH1 * 32;                 // 18*18*32
  _Float16* s_w2  = s_h2 + RH2 * RH2 * 32;                 // 9*32*32
  _Float16* s_w3  = s_w2 + 9 * 32 * 32;                    // 9*16*32
  float*    s_w1  = (float*)(s_w3 + 9 * 16 * 32);          // 27*32
  float*    s_b1  = s_w1 + 27 * 32;
  float*    s_b2  = s_b1 + 32;
  float*    s_b3  = s_b2 + 32;                             // 4 (pad)
  float*    s_gen = s_b3 + 4;                              // 16*16*3
  float*    s_red = s_gen + TILE * TILE * 3;               // 8

  // ---- stage weights into LDS (f16, B-layout transposed to [tap][n][k]) ----
  for (int i = tid; i < 9 * 32 * 32; i += 256) {
    int tap = i >> 10, n = (i >> 5) & 31, k = i & 31;
    s_w2[i] = (_Float16)W2[((size_t)(tap * 32 + k)) * 32 + n];     // HWIO [tap][k][n]
  }
  for (int i = tid; i < 9 * 16 * 32; i += 256) {
    int tap = i >> 9, n = (i >> 5) & 15, k = i & 31;
    s_w3[i] = (n < 3) ? (_Float16)W3[((size_t)(tap * 32 + k)) * 3 + n] : (_Float16)0.0f;
  }
  for (int i = tid; i < 27 * 32; i += 256) s_w1[i] = W1[i];        // [kin27][cout32]
  if (tid < 32) { s_b1[tid] = b1[tid]; s_b2[tid] = b2[tid]; }
  if (tid < 4)  s_b3[tid] = (tid < 3) ? b3[tid] : 0.0f;

  // ---- crop_resize the input region (zeros outside the 128x128 patch) ----
  const float* bx = wsBoxes + (size_t)(img * MB_ + step) * 4;
  const float bymin = bx[0], bxmin = bx[1], bymax = bx[2], bxmax = bx[3];
  const float bh = fmaxf(bymax - bymin, 1.0f);
  const float bw = fmaxf(bxmax - bxmin, 1.0f);
  const float* ib = images + (size_t)img * IMG_ * IMG_ * 3;

  for (int i = tid; i < RIN * RIN; i += 256) {
    int ri = i / RIN, rj = i % RIN;
    int pyi = r0 - 3 + ri, pxi = c0 - 3 + rj;            // patch coords
    float v0 = 0.f, v1 = 0.f, v2 = 0.f;
    if (pyi >= 0 && pyi < P_ && pxi >= 0 && pxi < P_) {
      float ys = bymin + ((float)pyi + 0.5f) * (1.0f / P_) * bh - 0.5f;
      float xs = bxmin + ((float)pxi + 0.5f) * (1.0f / P_) * bw - 0.5f;
      ys = fminf(fmaxf(ys, 0.0f), 511.0f);
      xs = fminf(fmaxf(xs, 0.0f), 511.0f);
      int y0i = (int)floorf(ys), x0i = (int)floorf(xs);
      int y1i = imin(y0i + 1, 511), x1i = imin(x0i + 1, 511);
      float wy = ys - (float)y0i, wx = xs - (float)x0i;
      const float* p00 = ib + ((size_t)y0i * IMG_ + x0i) * 3;
      const float* p01 = ib + ((size_t)y0i * IMG_ + x1i) * 3;
      const float* p10 = ib + ((size_t)y1i * IMG_ + x0i) * 3;
      const float* p11 = ib + ((size_t)y1i * IMG_ + x1i) * 3;
      v0 = (p00[0]*(1.f-wx)+p01[0]*wx)*(1.f-wy) + (p10[0]*(1.f-wx)+p11[0]*wx)*wy;
      v1 = (p00[1]*(1.f-wx)+p01[1]*wx)*(1.f-wy) + (p10[1]*(1.f-wx)+p11[1]*wx)*wy;
      v2 = (p00[2]*(1.f-wx)+p01[2]*wx)*(1.f-wy) + (p10[2]*(1.f-wx)+p11[2]*wx)*wy;
    }
    s_in[i * 3 + 0] = v0; s_in[i * 3 + 1] = v1; s_in[i * 3 + 2] = v2;
  }
  __syncthreads();

  // ---- conv1 (3 -> 32), scalar f32, relu, zero outside patch ----
  for (int it = tid; it < RH1 * RH1 * 4; it += 256) {
    int pix = it >> 2, ch0 = (it & 3) << 3;
    int ri = pix / RH1, rj = pix % RH1;
    int pyi = r0 - 2 + ri, pxi = c0 - 2 + rj;
    bool inP = (pyi >= 0) && (pyi < P_) && (pxi >= 0) && (pxi < P_);
    float acc[8];
    #pragma unroll
    for (int n = 0; n < 8; ++n) acc[n] = s_b1[ch0 + n];
    if (inP) {
      #pragma unroll
      for (int dy = 0; dy < 3; ++dy)
        #pragma unroll
        for (int dx = 0; dx < 3; ++dx) {
          const float* ip = &s_in[((ri + dy) * RIN + (rj + dx)) * 3];
          const float* wp = &s_w1[(dy * 3 + dx) * 3 * 32 + ch0];
          #pragma unroll
          for (int cin = 0; cin < 3; ++cin) {
            float x = ip[cin];
            #pragma unroll
            for (int n = 0; n < 8; ++n) acc[n] = fmaf(x, wp[cin * 32 + n], acc[n]);
          }
        }
    }
    #pragma unroll
    for (int n = 0; n < 8; ++n)
      s_h1[pix * 32 + ch0 + n] = inP ? (_Float16)fmaxf(acc[n], 0.0f) : (_Float16)0.0f;
  }
  __syncthreads();

  const int klo  = (lane < 16) ? 0 : 8;    // A-fragment K base (ISA 16-bit A 16x32 layout)
  const int bklo = (lane < 16) ? 0 : 8;    // B-fragment K base
  const int nn   = lane & 15;

  // ---- conv2 (32 -> 32) via WMMA: 21 pixel-groups x 2 N-halves ----
  for (int u = wave; u < 42; u += 8) {
    int g = u >> 1, n0 = (u & 1) << 4;
    int p = g * 16 + (lane & 15);
    if (p > RH2 * RH2 - 1) p = RH2 * RH2 - 1;
    int row = p / RH2, col = p % RH2;

    v8f acc;
    float bv = s_b2[n0 + nn];
    #pragma unroll
    for (int v = 0; v < 8; ++v) acc[v] = bv;

    for (int tap = 0; tap < 9; ++tap) {
      int dy = tap / 3, dx = tap % 3;
      const _Float16* ap = &s_h1[((row + dy) * RH1 + (col + dx)) * 32];
      v16h A;
      #pragma unroll
      for (int v = 0; v < 8; ++v) {
        int kk = ((v < 4) ? 2 * v : 8 + 2 * v) + klo;
        h2v pr = *(const h2v*)(ap + kk);
        A[2 * v] = pr[0]; A[2 * v + 1] = pr[1];
      }
      const _Float16* bp = &s_w2[(tap * 32 + n0 + nn) * 32];
      v16h Bf;
      #pragma unroll
      for (int v = 0; v < 8; ++v) {
        h2v pr = *(const h2v*)(bp + ((v < 4) ? 2 * v : 8 + 2 * v) + bklo);
        Bf[2 * v] = pr[0]; Bf[2 * v + 1] = pr[1];
      }
      acc = __builtin_amdgcn_wmma_f32_16x16x32_f16(false, A, false, Bf, (short)0, acc, false, false);
    }
    // relu + store h2 (zero-pad positions outside the 128x128 patch)
    #pragma unroll
    for (int v = 0; v < 8; ++v) {
      int mm = v + ((lane < 16) ? 0 : 8);
      int pp = g * 16 + mm;
      if (pp < RH2 * RH2) {
        int rr = pp / RH2, cc = pp % RH2;
        int pyi = r0 - 1 + rr, pxi = c0 - 1 + cc;
        bool inP = (pyi >= 0) && (pyi < P_) && (pxi >= 0) && (pxi < P_);
        s_h2[pp * 32 + n0 + nn] = inP ? (_Float16)fmaxf(acc[v], 0.0f) : (_Float16)0.0f;
      }
    }
  }
  __syncthreads();

  // ---- conv3 (32 -> 3, N padded to 16) via WMMA, tanh ----
  for (int g = wave; g < 16; g += 8) {
    int p = g * 16 + (lane & 15);
    int row = p >> 4, col = p & 15;

    v8f acc;
    float bv = (nn < 3) ? s_b3[nn] : 0.0f;
    #pragma unroll
    for (int v = 0; v < 8; ++v) acc[v] = bv;

    for (int tap = 0; tap < 9; ++tap) {
      int dy = tap / 3, dx = tap % 3;
      const _Float16* ap = &s_h2[((row + dy) * RH2 + (col + dx)) * 32];
      v16h A;
      #pragma unroll
      for (int v = 0; v < 8; ++v) {
        int kk = ((v < 4) ? 2 * v : 8 + 2 * v) + klo;
        h2v pr = *(const h2v*)(ap + kk);
        A[2 * v] = pr[0]; A[2 * v + 1] = pr[1];
      }
      const _Float16* bp = &s_w3[(tap * 16 + nn) * 32];
      v16h Bf;
      #pragma unroll
      for (int v = 0; v < 8; ++v) {
        h2v pr = *(const h2v*)(bp + ((v < 4) ? 2 * v : 8 + 2 * v) + bklo);
        Bf[2 * v] = pr[0]; Bf[2 * v + 1] = pr[1];
      }
      acc = __builtin_amdgcn_wmma_f32_16x16x32_f16(false, A, false, Bf, (short)0, acc, false, false);
    }
    if (nn < 3) {
      #pragma unroll
      for (int v = 0; v < 8; ++v) {
        int mm = v + ((lane < 16) ? 0 : 8);
        int pp = g * 16 + mm;
        s_gen[pp * 3 + nn] = tanhf(acc[v]);
      }
    }
  }

  if (tid == 0) s_red[0] = 0.0f;
  __syncthreads();

  // ---- MSE partial + write generated tile to global gen buffer ----
  float lsum = 0.0f;
  for (int i = tid; i < TILE * TILE * 3; i += 256) {
    int pix = i / 3, c = i % 3;
    int rr = pix >> 4, cc = pix & 15;
    float gv = s_gen[i];
    float pv = s_in[((rr + 3) * RIN + (cc + 3)) * 3 + c];
    float d = gv - pv;
    lsum += d * d;
    genbuf[(((size_t)img * P_ + (r0 + rr)) * P_ + (c0 + cc)) * 3 + c] = gv;
  }
  atomicAdd(&s_red[0], lsum);
  __syncthreads();
  if (tid == 0) atomicAdd(&lossacc[img], s_red[0] * (1.0f / (P_ * P_ * 3)));
}

// ---------------------------------------------------------------------------
// Paste: bilinear-resample gen patch back into image, clip. grid=(1024, B).
// ---------------------------------------------------------------------------
__global__ __launch_bounds__(256) void paste_kernel(float* __restrict__ images,
                                                    const float* __restrict__ genbuf,
                                                    const float* __restrict__ wsBoxes,
                                                    const int* __restrict__ wsValid,
                                                    int step) {
  const int img = blockIdx.y;
  if (!wsValid[img * MB_ + step]) return;
  int idx = blockIdx.x * 256 + threadIdx.x;
  int yy = idx >> 9, xx = idx & 511;

  const float* bx = wsBoxes + (size_t)(img * MB_ + step) * 4;
  float ymin = bx[0], xmin = bx[1], ymax = bx[2], xmax = bx[3];
  float fy = (float)yy, fx = (float)xx;
  if (!(fy >= ymin && fy < ymax && fx >= xmin && fx < xmax)) return;

  float h = fmaxf(ymax - ymin, 1.0f), w = fmaxf(xmax - xmin, 1.0f);
  float py = (fy - ymin) / h * (float)P_ - 0.5f;
  float px = (fx - xmin) / w * (float)P_ - 0.5f;
  py = fminf(fmaxf(py, 0.0f), 127.0f);
  px = fminf(fmaxf(px, 0.0f), 127.0f);
  int y0 = (int)floorf(py), x0 = (int)floorf(px);
  int y1 = imin(y0 + 1, 127), x1 = imin(x0 + 1, 127);
  float wy = py - (float)y0, wx = px - (float)x0;

  const float* gb = genbuf + (size_t)img * P_ * P_ * 3;
  float* op = images + ((size_t)img * IMG_ * IMG_ + (size_t)yy * IMG_ + xx) * 3;
  #pragma unroll
  for (int c = 0; c < 3; ++c) {
    float v = (gb[(y0 * P_ + x0) * 3 + c] * (1.f - wx) + gb[(y0 * P_ + x1) * 3 + c] * wx) * (1.f - wy)
            + (gb[(y1 * P_ + x0) * 3 + c] * (1.f - wx) + gb[(y1 * P_ + x1) * 3 + c] * wx) * wy;
    op[c] = fminf(fmaxf(v, -1.0f), 1.0f);
  }
}

// ---------------------------------------------------------------------------
// Final loss: sum_b (max_score_b^2 + mse_loss_b)
// ---------------------------------------------------------------------------
__global__ void final_kernel(const float* __restrict__ wsMaxSc,
                             const float* __restrict__ wsLoss,
                             float* __restrict__ out) {
  if (threadIdx.x == 0 && blockIdx.x == 0) {
    float l = 0.0f;
    for (int b = 0; b < B_; ++b) l += wsMaxSc[b] * wsMaxSc[b] + wsLoss[b];
    out[0] = l;
  }
}

// ---------------------------------------------------------------------------
extern "C" void kernel_launch(void* const* d_in, const int* in_sizes, int n_in,
                              void* d_out, int out_size, void* d_ws, size_t ws_size,
                              hipStream_t stream) {
  (void)in_sizes; (void)n_in; (void)out_size; (void)ws_size;
  const float* images      = (const float*)d_in[0];
  const float* cand_boxes  = (const float*)d_in[1];
  const float* cand_scores = (const float*)d_in[2];
  const float* W1 = (const float*)d_in[3];
  const float* b1 = (const float*)d_in[4];
  const float* W2 = (const float*)d_in[5];
  const float* b2 = (const float*)d_in[6];
  const float* W3 = (const float*)d_in[7];
  const float* b3 = (const float*)d_in[8];

  float* out_img = (float*)d_out;

  // workspace layout (floats): boxes[8*16*4] | valid[8*16] | maxsc[8] | loss[8] | genbuf[8*128*128*3]
  float* ws       = (float*)d_ws;
  float* wsBoxes  = ws;                     // 512
  int*   wsValid  = (int*)(ws + 512);       // 128
  float* wsMaxSc  = ws + 640;               // 8
  float* wsLoss   = ws + 648;               // 8
  float* genbuf   = ws + 656;               // 393216

  const int nImg = B_ * IMG_ * IMG_ * 3;

  copy_kernel<<<(nImg + 255) / 256, 256, 0, stream>>>(images, out_img, nImg);
  nms_kernel<<<B_, 256, 0, stream>>>(cand_boxes, cand_scores, wsBoxes, wsValid, wsMaxSc, wsLoss);

  for (int step = 0; step < MB_; ++step) {
    gen_kernel<<<dim3(64, B_), 256, GEN_SMEM_BYTES, stream>>>(
        out_img, wsBoxes, wsValid, W1, b1, W2, b2, W3, b3, genbuf, wsLoss, step);
    paste_kernel<<<dim3((IMG_ * IMG_) / 256, B_), 256, 0, stream>>>(
        out_img, genbuf, wsBoxes, wsValid, step);
  }

  final_kernel<<<1, 32, 0, stream>>>(wsMaxSc, wsLoss, out_img + nImg);
}
